// self_attention_19018115187303
// MI455X (gfx1250) — matrix operand order repaired and checked by reading, not compiled, not run on hardware
//
#include <hip/hip_runtime.h>

typedef _Float16 half_t;
typedef __attribute__((ext_vector_type(16))) _Float16 v16h;
typedef __attribute__((ext_vector_type(8)))  _Float16 v8h;
typedef __attribute__((ext_vector_type(8)))  float    v8f;
typedef __attribute__((ext_vector_type(4)))  float    v4f;

#define LDH   136   // padded f16 leading dim (128 + 8): b128 alignment + bank spread
#define LDS_S 132   // padded f32 leading dim
#define LDO   264   // padded f16 leading dim for 256-wide staging tile

// A fragment (16x32 f16, M x K), source row-major with leading dim `ld`.
// Lane m=lane&15, g=lane>>4; VGPR0-3 hold K = g*8 + 0..7,
// VGPR4-7 hold K = 16 + g*8 + 0..7  -> two contiguous 16B chunks.
__device__ __forceinline__ v16h frag_a16(const half_t* base, int row0, int k0,
                                         int ld, int lane) {
  const int m = lane & 15, g = lane >> 4;
  const half_t* p = base + (row0 + m) * ld + k0 + g * 8;
  v8h lo = *(const v8h*)p;
  v8h hi = *(const v8h*)(p + 16);
  v16h r;
#pragma unroll
  for (int i = 0; i < 8; ++i) { r[i] = lo[i]; r[i + 8] = hi[i]; }
  return r;
}

// B fragment (32x16 f16, K x N) where B[k][n] = S[n0+n][k0+k], S row-major
// with leading dim `ld` (contraction contiguous). Lane n=lane&15, g=lane>>4
// holds K = g*8 + 0..7 and K = 16 + g*8 + 0..7.
__device__ __forceinline__ v16h frag_bT16(const half_t* S, int n0, int k0,
                                          int ld, int lane) {
  const int n = lane & 15, g = lane >> 4;
  const half_t* p = S + (n0 + n) * ld + k0 + g * 8;
  v8h lo = *(const v8h*)p;
  v8h hi = *(const v8h*)(p + 16);
  v16h r;
#pragma unroll
  for (int i = 0; i < 8; ++i) { r[i] = lo[i]; r[i + 8] = hi[i]; }
  return r;
}

// ---------------------------------------------------------------------------
// Kernel 1: QKV projection.  out[e,p] = sum_c W[e,c] * x[b,c,p] + bias[e]
// Grid: 16 batches x 64 pixel-tiles (256 pixels).  256 threads = 8 waves.
// ---------------------------------------------------------------------------
__global__ __launch_bounds__(256)
void qkv_proj_kernel(const float* __restrict__ x,
                     const float* __restrict__ wq, const float* __restrict__ bq,
                     const float* __restrict__ wk, const float* __restrict__ bk,
                     const float* __restrict__ wv, const float* __restrict__ bv,
                     half_t* __restrict__ qws, half_t* __restrict__ kws,
                     half_t* __restrict__ vws) {
  __shared__ half_t shW[3 * 64 * 72];  // [q][e][c] f16, ld 72
  __shared__ half_t shX[256 * 72];     // [p][c]    f16, ld 72 (transposed x tile)
  __shared__ half_t shO[64 * LDO];     // staging tile [e][p] f16
  __shared__ float  shBias[192];

  const int t  = threadIdx.x;
  const int b  = blockIdx.x >> 6;
  const int p0 = (blockIdx.x & 63) * 256;

  for (int idx = t; idx < 3 * 4096; idx += 256) {
    const int q = idx >> 12, r = idx & 4095;
    const float* w = (q == 0) ? wq : ((q == 1) ? wk : wv);
    shW[q * 4608 + (r >> 6) * 72 + (r & 63)] = (half_t)w[r];
  }
  if (t < 192) {
    const float* bb = (t < 64) ? bq : ((t < 128) ? bk : bv);
    shBias[t] = bb[t & 63];
  }
  const float* xb = x + (size_t)b * 64 * 16384 + p0;
  for (int idx = t; idx < 64 * 256; idx += 256) {
    const int c = idx >> 8, i = idx & 255;
    shX[i * 72 + c] = (half_t)xb[(size_t)c * 16384 + i];
  }
  __syncthreads();

  const int wv_id = t >> 5, lane = t & 31;
  const int n = lane & 15, g = lane >> 4;

  // Hoist B fragments (x tile) for this wave's two 16-wide pixel tiles.
  v16h bf[2][2];
#pragma unroll
  for (int ns = 0; ns < 2; ++ns) {
    const int nt = wv_id * 2 + ns;
    bf[ns][0] = frag_bT16(shX, nt * 16, 0, 72, lane);
    bf[ns][1] = frag_bT16(shX, nt * 16, 32, 72, lane);
  }

  for (int q = 0; q < 3; ++q) {
    const half_t* W = shW + q * 4608;
#pragma unroll
    for (int mt = 0; mt < 4; ++mt) {
      const v16h a0 = frag_a16(W, mt * 16, 0, 72, lane);
      const v16h a1 = frag_a16(W, mt * 16, 32, 72, lane);
#pragma unroll
      for (int ns = 0; ns < 2; ++ns) {
        v8f c = {};
        c = __builtin_amdgcn_wmma_f32_16x16x32_f16(false, a0, false, bf[ns][0],
                                                   (short)0, c, false, false);
        c = __builtin_amdgcn_wmma_f32_16x16x32_f16(false, a1, false, bf[ns][1],
                                                   (short)0, c, false, false);
        const int pcol = wv_id * 32 + ns * 16 + n;
#pragma unroll
        for (int j = 0; j < 8; ++j) {
          const int e = mt * 16 + j + g * 8;
          shO[e * LDO + pcol] = (half_t)(c[j] + shBias[q * 64 + e]);
        }
      }
    }
    __syncthreads();
    // Cooperative, coalesced b128 store of the 64x256 tile.
    half_t* op = (q == 0) ? qws : ((q == 1) ? kws : vws);
    for (int c8 = t; c8 < 2048; c8 += 256) {
      const int idx = c8 * 8;
      const int e = idx >> 8, pp = idx & 255;
      const v8h val = *(const v8h*)&shO[e * LDO + pp];
      *(v8h*)(op + (size_t)(b * 64 + e) * 16384 + p0 + pp) = val;
    }
    __syncthreads();  // shO reused next q
  }
}

// ---------------------------------------------------------------------------
// Kernel 2: per-(b,e) attention over a 128x128 slice.  Grid: 1024 blocks.
// ---------------------------------------------------------------------------
__global__ __launch_bounds__(256)
void attn_kernel(const half_t* __restrict__ qws, const half_t* __restrict__ kws,
                 const half_t* __restrict__ vws, float* __restrict__ out) {
  __shared__ half_t shA[128 * LDH];   // Q [h][w]      -> temp1 [h][k]
  __shared__ half_t shB[128 * LDH];   // K [k][w]      -> temp2^T [v][w]
  __shared__ half_t shAt[128 * LDH];  // Q^T [w][h]    -> tv [h][w]
  __shared__ half_t shBt[128 * LDH];  // K^T [v][h]
  __shared__ half_t shVt[128 * LDH];  // V^T [w][k]
  __shared__ float  shS[128 * LDS_S]; // f32 score scratch / output staging

  const int t = threadIdx.x;
  const size_t base = (size_t)blockIdx.x * 16384;  // (b*64+e)*H*W
  const int wv_id = t >> 5, lane = t & 31;
  const int n = lane & 15, g = lane >> 4;

  // Stage Q, K (plain + transposed) and V^T in LDS (f16).
  for (int c8 = t; c8 < 2048; c8 += 256) {
    const int p0 = c8 * 8;
    const int h = p0 >> 7, w0 = p0 & 127;
    const v8h q = *(const v8h*)(qws + base + p0);
    const v8h k = *(const v8h*)(kws + base + p0);
    const v8h v = *(const v8h*)(vws + base + p0);
    *(v8h*)&shA[h * LDH + w0] = q;
    *(v8h*)&shB[h * LDH + w0] = k;
#pragma unroll
    for (int j = 0; j < 8; ++j) {
      shAt[(w0 + j) * LDH + h] = q[j];
      shBt[(w0 + j) * LDH + h] = k[j];
      shVt[(w0 + j) * LDH + h] = v[j];
    }
  }
  __syncthreads();

  // --- scores[h,k] = (1/8) * sum_w Q[h,w] K[k,w]  -> shS (f32) -------------
  {
    const int mt = wv_id;
    v16h a[4];
#pragma unroll
    for (int ks = 0; ks < 4; ++ks) a[ks] = frag_a16(shA, mt * 16, ks * 32, LDH, lane);
    for (int ntile = 0; ntile < 8; ++ntile) {
      v8f c = {};
#pragma unroll
      for (int ks = 0; ks < 4; ++ks) {
        const v16h bfr = frag_bT16(shB, ntile * 16, ks * 32, LDH, lane);
        c = __builtin_amdgcn_wmma_f32_16x16x32_f16(false, a[ks], false, bfr,
                                                   (short)0, c, false, false);
      }
#pragma unroll
      for (int j = 0; j < 8; ++j)
        shS[(mt * 16 + j + g * 8) * LDS_S + ntile * 16 + n] = c[j] * 0.125f;
    }
  }
  __syncthreads();

  // --- row softmax of shS -> temp1 (f16) into shA --------------------------
  if (t < 128) {
    float* row = shS + t * LDS_S;
    float mx = -1e30f;
    for (int i = 0; i < 128; ++i) mx = fmaxf(mx, row[i]);
    float sum = 0.f;
    for (int i = 0; i < 128; ++i) { const float e = __expf(row[i] - mx); row[i] = e; sum += e; }
    const float inv = 1.f / sum;
    half_t* dst = shA + t * LDH;
    for (int i = 0; i < 128; ++i) dst[i] = (half_t)(row[i] * inv);
  }
  __syncthreads();

  // --- t2[w,v] = (1/8) * sum_h Q[h,w] K[h,v]  -> shS (f32) -----------------
  {
    const int mt = wv_id;
    v16h a[4];
#pragma unroll
    for (int ks = 0; ks < 4; ++ks) a[ks] = frag_a16(shAt, mt * 16, ks * 32, LDH, lane);
    for (int ntile = 0; ntile < 8; ++ntile) {
      v8f c = {};
#pragma unroll
      for (int ks = 0; ks < 4; ++ks) {
        const v16h bfr = frag_bT16(shBt, ntile * 16, ks * 32, LDH, lane);
        c = __builtin_amdgcn_wmma_f32_16x16x32_f16(false, a[ks], false, bfr,
                                                   (short)0, c, false, false);
      }
#pragma unroll
      for (int j = 0; j < 8; ++j)
        shS[(mt * 16 + j + g * 8) * LDS_S + ntile * 16 + n] = c[j] * 0.125f;
    }
  }
  __syncthreads();

  // --- column softmax (over w) -> temp2^T (f16) into shB[v][w] -------------
  if (t < 128) {
    float mx = -1e30f;
    for (int i = 0; i < 128; ++i) mx = fmaxf(mx, shS[i * LDS_S + t]);
    float sum = 0.f;
    for (int i = 0; i < 128; ++i) sum += __expf(shS[i * LDS_S + t] - mx);
    const float inv = 1.f / sum;
    half_t* dst = shB + t * LDH;
    for (int i = 0; i < 128; ++i)
      dst[i] = (half_t)(__expf(shS[i * LDS_S + t] - mx) * inv);
  }
  __syncthreads();

  // --- tv[h,w] = sum_k temp1[h,k] V[k,w]  -> f16 into shAt ----------------
  {
    const int mt = wv_id;
    v16h a[4];
#pragma unroll
    for (int ks = 0; ks < 4; ++ks) a[ks] = frag_a16(shA, mt * 16, ks * 32, LDH, lane);
    for (int ntile = 0; ntile < 8; ++ntile) {
      v8f c = {};
#pragma unroll
      for (int ks = 0; ks < 4; ++ks) {
        const v16h bfr = frag_bT16(shVt, ntile * 16, ks * 32, LDH, lane);
        c = __builtin_amdgcn_wmma_f32_16x16x32_f16(false, a[ks], false, bfr,
                                                   (short)0, c, false, false);
      }
#pragma unroll
      for (int j = 0; j < 8; ++j)
        shAt[(mt * 16 + j + g * 8) * LDH + ntile * 16 + n] = (half_t)c[j];
    }
  }
  __syncthreads();

  // --- out[h,v] = sum_w tv[h,w] temp2[w,v]  -> shS, then coalesced store ---
  {
    const int mt = wv_id;
    v16h a[4];
#pragma unroll
    for (int ks = 0; ks < 4; ++ks) a[ks] = frag_a16(shAt, mt * 16, ks * 32, LDH, lane);
    for (int ntile = 0; ntile < 8; ++ntile) {
      v8f c = {};
#pragma unroll
      for (int ks = 0; ks < 4; ++ks) {
        const v16h bfr = frag_bT16(shB, ntile * 16, ks * 32, LDH, lane);
        c = __builtin_amdgcn_wmma_f32_16x16x32_f16(false, a[ks], false, bfr,
                                                   (short)0, c, false, false);
      }
#pragma unroll
      for (int j = 0; j < 8; ++j)
        shS[(mt * 16 + j + g * 8) * LDS_S + ntile * 16 + n] = c[j];
    }
  }
  __syncthreads();

  // Cooperative, coalesced float4 store of the 128x128 f32 result.
  {
    float* gout = out + base;
    for (int c4 = t; c4 < 4096; c4 += 256) {
      const int idx = c4 * 4;
      const int row = idx >> 7, col = idx & 127;
      const v4f val = *(const v4f*)&shS[row * LDS_S + col];
      *(v4f*)(gout + idx) = val;
    }
  }
}

extern "C" void kernel_launch(void* const* d_in, const int* in_sizes, int n_in,
                              void* d_out, int out_size, void* d_ws, size_t ws_size,
                              hipStream_t stream) {
  const float* x  = (const float*)d_in[0];
  const float* wq = (const float*)d_in[1];
  const float* bq = (const float*)d_in[2];
  const float* wk = (const float*)d_in[3];
  const float* bk = (const float*)d_in[4];
  const float* wv = (const float*)d_in[5];
  const float* bv = (const float*)d_in[6];
  float* out = (float*)d_out;

  const size_t per = (size_t)16 * 64 * 16384;  // B*E*H*W halves per tensor
  half_t* qws = (half_t*)d_ws;
  half_t* kws = qws + per;
  half_t* vws = kws + per;

  qkv_proj_kernel<<<1024, 256, 0, stream>>>(x, wq, bq, wk, bk, wv, bv,
                                            qws, kws, vws);
  attn_kernel<<<1024, 256, 0, stream>>>(qws, kws, vws, out);
}
